// Warping_85950885527865
// MI455X (gfx1250) — compile-verified, matrix operand order checked by reading, not run on hardware
//
#include <hip/hip_runtime.h>

typedef __attribute__((ext_vector_type(16))) _Float16 v16h;
typedef __attribute__((ext_vector_type(8)))  float    v8f;
typedef __attribute__((ext_vector_type(4)))  unsigned v4u;
typedef __attribute__((ext_vector_type(4)))  int      v4i;
typedef __attribute__((ext_vector_type(8)))  int      v8i;
typedef _Float16 half_t;

// Problem constants (from reference)
constexpr int Bn   = 16;
constexpr int C0   = 26;
constexpr int Hh   = 96;
constexpr int Ww   = 72;
constexpr int HW   = Hh * Ww;
constexpr int P    = Bn * HW;       // 110592, divisible by 128 and 256
constexpr int OFFC   = C0 * 2 * 9;  // 468
constexpr int OFFPAD = 480;         // 468 padded to multiple of 16

#if defined(__gfx1250__) && __has_builtin(__builtin_amdgcn_tensor_load_to_lds)
#define HAVE_TDM 1
#else
#define HAVE_TDM 0
#endif

#if HAVE_TDM
// ---------------------------------------------------------------------------
// Issue one 2-D TDM tile load: rows x cpadHalves f16 elements, row stride
// rowStrideHalves, global tile start gsrc -> LDS byte address ldsAddr.
// D# layout per CDNA5 ISA 8.3/8.4 (group0 128b, group1 256b; groups 2/3 zero).
// ---------------------------------------------------------------------------
__device__ __forceinline__ void tdm_load_tile(const half_t* gsrc, unsigned ldsAddr,
                                              unsigned cpadHalves, unsigned validRows,
                                              unsigned rowStrideHalves)
{
    const unsigned long long ga = (unsigned long long)(const void*)gsrc;
    v4u g0;
    g0[0] = 1u;                                           // count=1, user mode
    g0[1] = ldsAddr;                                      // lds_addr
    g0[2] = (unsigned)(ga & 0xffffffffu);                 // global_addr[31:0]
    g0[3] = (unsigned)((ga >> 32) & 0x01ffffffu)          // global_addr[56:32]
            | 0x80000000u;                                // type=2 ("image")
    v8i g1;
    g1[0] = (int)(1u << 16);                              // data_size=1 -> 2 bytes
    g1[1] = (int)((cpadHalves & 0xffffu) << 16);          // tensor_dim0[15:0]
    g1[2] = (int)((cpadHalves >> 16) | ((validRows & 0xffffu) << 16)); // dim0 hi | dim1 lo
    g1[3] = (int)((validRows >> 16) | (cpadHalves << 16));             // dim1 hi | tile_dim0
    g1[4] = (int)128u;                                    // tile_dim1=128, tile_dim2=0
    g1[5] = (int)rowStrideHalves;                         // tensor_dim0_stride[31:0]
    g1[6] = 0;                                            // stride0 hi | stride1 lo
    g1[7] = 0;                                            // stride1 hi
    v4i gz = {0, 0, 0, 0};
#if __has_include(<hip/amd_detail/amd_gfx1250_TDM.h>)
    v8i gz8 = {0, 0, 0, 0, 0, 0, 0, 0};
    __builtin_amdgcn_tensor_load_to_lds(g0, g1, gz, gz, gz8, 0);
#else
    __builtin_amdgcn_tensor_load_to_lds(g0, g1, gz, gz, 0);
#endif
}
#endif  // HAVE_TDM

// ---------------------------------------------------------------------------
// Fused implicit-GEMM 3x3 (or 1x1) conv + BN + residual + ReLU, f16 WMMA.
// in : NHWC f16, channel stride CPAD
// wt : [Npad][ntaps][CPAD] f16 (pre-transformed, zero padded)
// out: NHWC f16, channel stride outStride
// Workgroup: 8 waves -> 128 M rows x 128 N cols.  pad == dil assumed.
// Weight tile per tap staged to double-buffered LDS by the Tensor Data Mover;
// DMA of tap t+1 overlaps the WMMAs of tap t.
// ---------------------------------------------------------------------------
template <int CPAD>
__global__ __launch_bounds__(256, 4)
void conv_wmma(const half_t* __restrict__ in, const half_t* __restrict__ wt,
               const float* __restrict__ scale, const float* __restrict__ shift,
               const half_t* __restrict__ res, half_t* __restrict__ out,
               int Npad, int outStride, int ntaps, int dil, int relu)
{
    __shared__ half_t wlds[2][128 * CPAD];

    const int tid   = threadIdx.x;
    const int lane  = tid & 31;
    const int wave  = tid >> 5;
    const int mBase = blockIdx.x * 128;
    const int nBase = blockIdx.y * 128;
    int ntiles = (Npad - nBase) >> 4;
    if (ntiles > 8) ntiles = 8;

    // A-fragment lane mapping (16-bit A 16x32): row = lane&15, K half = lane>>4
    const int row   = lane & 15;
    const int khalf = lane >> 4;
    const int m     = mBase + wave * 16 + row;
    const int bIdx  = m / HW;
    const int rem   = m % HW;
    const int y     = rem / Ww;
    const int x     = rem % Ww;

    v8f zero8 = {};
    v8f acc[8];
#pragma unroll
    for (int t = 0; t < 8; ++t) acc[t] = zero8;

    constexpr unsigned TILE_BYTES = 128u * CPAD * 2u;

#if HAVE_TDM
    const unsigned ldsBase = (unsigned)(unsigned long long)(const void*)&wlds[0][0];
    if (wave == 0)
        tdm_load_tile(wt + (size_t)nBase * ntaps * CPAD, ldsBase,
                      (unsigned)CPAD, (unsigned)(Npad - nBase), (unsigned)(ntaps * CPAD));
#else
    constexpr int F4_PER_ROW = CPAD / 8;
#endif

    for (int tap = 0; tap < ntaps; ++tap) {
#if HAVE_TDM
        if (wave == 0) {
            if (tap + 1 < ntaps) {
                tdm_load_tile(wt + ((size_t)nBase * ntaps + tap + 1) * CPAD,
                              ldsBase + (unsigned)((tap + 1) & 1) * TILE_BYTES,
                              (unsigned)CPAD, (unsigned)(Npad - nBase),
                              (unsigned)(ntaps * CPAD));
                __builtin_amdgcn_s_wait_tensorcnt(1);   // tap's tile has landed
            } else {
                __builtin_amdgcn_s_wait_tensorcnt(0);
            }
        }
        __syncthreads();
#else
        __syncthreads();
        for (int f = tid; f < 128 * F4_PER_ROW; f += 256) {
            const int n  = f / F4_PER_ROW;
            const int kk = f - n * F4_PER_ROW;
            const float4* src = reinterpret_cast<const float4*>(
                wt + ((size_t)(nBase + n) * ntaps + tap) * CPAD + kk * 8);
            reinterpret_cast<float4*>(&wlds[tap & 1][n * CPAD])[kk] = *src;
        }
        __syncthreads();
#endif
        const half_t* wbuf = &wlds[tap & 1][0];

        int dy = 0, dx = 0;
        if (ntaps == 9) { dy = (tap / 3 - 1) * dil; dx = (tap % 3 - 1) * dil; }
        const int yy = y + dy;
        const int xx = x + dx;
        const bool inb = (yy >= 0) && (yy < Hh) && (xx >= 0) && (xx < Ww);

#pragma unroll
        for (int kc = 0; kc < CPAD; kc += 32) {
            v16h afrag = {};
            if (inb) {
                const float4* ap = reinterpret_cast<const float4*>(
                    in + ((size_t)((bIdx * Hh + yy) * Ww + xx)) * CPAD + kc + khalf * 16);
                reinterpret_cast<float4*>(&afrag)[0] = ap[0];
                reinterpret_cast<float4*>(&afrag)[1] = ap[1];
            }
#pragma unroll
            for (int t = 0; t < 8; ++t) {
                if (t < ntiles) {
                    // B-fragment (32x16): lane col = lane&15, 16 contiguous K halves
                    const float4* bp = reinterpret_cast<const float4*>(
                        wbuf + (t * 16 + row) * CPAD + kc + khalf * 16);
                    v16h bfrag;
                    reinterpret_cast<float4*>(&bfrag)[0] = bp[0];
                    reinterpret_cast<float4*>(&bfrag)[1] = bp[1];
                    acc[t] = __builtin_amdgcn_wmma_f32_16x16x32_f16(
                        false, afrag, false, bfrag, (short)0, acc[t], false, false);
                }
            }
        }
#if HAVE_TDM
        __syncthreads();   // all waves done reading buf[tap&1] before it is re-filled
#endif
    }

    // Epilogue: folded BN, residual add, ReLU, store f16.
    // C/D layout: element (vr, lane) -> M = vr + 8*(lane>>4), N = lane&15
#pragma unroll
    for (int t = 0; t < 8; ++t) {
        if (t >= ntiles) continue;
        const int n  = nBase + t * 16 + (lane & 15);
        const float sc = scale ? scale[n] : 1.0f;
        const float sh = scale ? shift[n] : 0.0f;
#pragma unroll
        for (int r = 0; r < 8; ++r) {
            const int mrow = mBase + wave * 16 + r + 8 * khalf;
            float v = acc[t][r] * sc + sh;
            if (res)  v += (float)res[(size_t)mrow * 128 + n];  // residuals always 128-ch
            if (relu) v = fmaxf(v, 0.0f);
            out[(size_t)mrow * outStride + n] = (half_t)v;
        }
    }
}

// ---------------------------------------------------------------------------
// Input prep: (ref - sup) -> NHWC f16 padded to 32 ch; sup -> NHWC f32 (26 ch)
// ---------------------------------------------------------------------------
__global__ void prep_inputs(const float* __restrict__ ref, const float* __restrict__ sup,
                            half_t* __restrict__ diff32, float* __restrict__ supN)
{
    const int t = blockIdx.x * blockDim.x + threadIdx.x;
    if (t >= P * 32) return;
    const int c = t & 31;
    const int p = t >> 5;
    const int b = p / HW;
    const int rem = p % HW;
    const int y = rem / Ww;
    const int x = rem % Ww;
    float d = 0.0f;
    if (c < C0) {
        const size_t idx = ((size_t)(b * C0 + c) * Hh + y) * Ww + x;
        const float r = ref[idx];
        const float s = sup[idx];
        d = r - s;
        supN[(size_t)p * C0 + c] = s;
    }
    diff32[(size_t)p * 32 + c] = (half_t)d;
}

// ---------------------------------------------------------------------------
// Weight transform: OIHW f32 -> [Npad][taps][Cpad] f16 (zero padded)
// ---------------------------------------------------------------------------
__global__ void wt_transform(const float* __restrict__ w, half_t* __restrict__ wt,
                             int O, int I, int taps, int Npad, int Cpad)
{
    const int t = blockIdx.x * blockDim.x + threadIdx.x;
    if (t >= Npad * taps * Cpad) return;
    const int c   = t % Cpad;
    const int tmp = t / Cpad;
    const int tap = tmp % taps;
    const int o   = tmp / taps;
    float v = 0.0f;
    if (o < O && c < I) v = w[((size_t)o * I + c) * taps + tap];
    wt[t] = (half_t)v;
}

// BN fold: p = [4][C] rows (scale, bias, mean, var) -> per-channel scale/shift
__global__ void bn_fold(const float* __restrict__ p, float* __restrict__ sc,
                        float* __restrict__ sh, int C)
{
    const int i = blockIdx.x * blockDim.x + threadIdx.x;
    if (i >= C) return;
    const float inv = p[i] * rsqrtf(p[3 * C + i] + 1e-5f);
    sc[i] = inv;
    sh[i] = p[C + i] - p[2 * C + i] * inv;
}

__global__ void zero_f32(float* __restrict__ o, int n)
{
    const int i = blockIdx.x * blockDim.x + threadIdx.x;
    if (i < n) o[i] = 0.0f;
}

// ---------------------------------------------------------------------------
// Deformable conv branch (dg = 26, cpg = 1): bilinear sample sup per
// (channel, tap) with per-channel offsets, then 26x234 mat-vec; += 0.2 * res.
// ---------------------------------------------------------------------------
__global__ __launch_bounds__(256)
void deform_kernel(const float* __restrict__ supN, const half_t* __restrict__ off,
                   const float* __restrict__ w, float* __restrict__ out, int dil)
{
    __shared__ float wl[26 * 26 * 9];
    for (int i = threadIdx.x; i < 26 * 26 * 9; i += 256) wl[i] = w[i];
    __syncthreads();

    const int p = blockIdx.x * 256 + threadIdx.x;
    if (p >= P) return;
    const int b = p / HW;
    const int rem = p % HW;
    const int y = rem / Ww;
    const int x = rem % Ww;

    float acc[26];
#pragma unroll
    for (int o = 0; o < 26; ++o) acc[o] = 0.0f;

    const half_t* offp = off + (size_t)p * OFFPAD;

    for (int c = 0; c < 26; ++c) {
#pragma unroll
        for (int k = 0; k < 9; ++k) {
            const float dy = (float)offp[c * 18 + k * 2 + 0];
            const float dx = (float)offp[c * 18 + k * 2 + 1];
            const float py = (float)(y + (k / 3 - 1) * dil) + dy;
            const float px = (float)(x + (k % 3 - 1) * dil) + dx;
            const float y0f = floorf(py);
            const float x0f = floorf(px);
            const float ly = py - y0f;
            const float lx = px - x0f;
            const int y0 = (int)y0f;
            const int x0 = (int)x0f;
            float v = 0.0f;
#pragma unroll
            for (int dyi = 0; dyi < 2; ++dyi) {
#pragma unroll
                for (int dxi = 0; dxi < 2; ++dxi) {
                    const int yi = y0 + dyi;
                    const int xi = x0 + dxi;
                    const bool ok = (yi >= 0) && (yi < Hh) && (xi >= 0) && (xi < Ww);
                    const int yc = min(max(yi, 0), Hh - 1);
                    const int xc = min(max(xi, 0), Ww - 1);
                    const float g = ok ? supN[((size_t)((b * Hh + yc) * Ww + xc)) * C0 + c] : 0.0f;
                    const float wb = (dyi ? ly : 1.0f - ly) * (dxi ? lx : 1.0f - lx);
                    v += g * wb;
                }
            }
#pragma unroll
            for (int o = 0; o < 26; ++o) acc[o] += v * wl[(o * 26 + c) * 9 + k];
        }
    }
#pragma unroll
    for (int o = 0; o < 26; ++o) {
        const size_t idx = ((size_t)(b * C0 + o) * Hh + y) * Ww + x;
        out[idx] += acc[o] * 0.2f;
    }
}

// ---------------------------------------------------------------------------
extern "C" void kernel_launch(void* const* d_in, const int* in_sizes, int n_in,
                              void* d_out, int out_size, void* d_ws, size_t ws_size,
                              hipStream_t stream)
{
    const float* ref      = (const float*)d_in[0];
    const float* sup      = (const float*)d_in[1];
    const float* w1_0     = (const float*)d_in[2];
    const float* w2_0     = (const float*)d_in[3];
    const float* wd_0     = (const float*)d_in[4];
    const float* bn1_0    = (const float*)d_in[5];
    const float* bn2_0    = (const float*)d_in[6];
    const float* bnd_0    = (const float*)d_in[7];
    const float* w1_rest  = (const float*)d_in[8];
    const float* w2_rest  = (const float*)d_in[9];
    const float* bn1_rest = (const float*)d_in[10];
    const float* bn2_rest = (const float*)d_in[11];
    const float* off_ws   = (const float*)d_in[12];
    const float* def_ws   = (const float*)d_in[13];

    // Workspace carve-out
    char* ws = (char*)d_ws;
    auto alloc = [&](size_t bytes) {
        char* r = ws;
        ws += (bytes + 255) & ~(size_t)255;
        return r;
    };
    half_t* diff32 = (half_t*)alloc((size_t)P * 32 * sizeof(half_t));
    half_t* bufs[3];
    for (int i = 0; i < 3; ++i) bufs[i] = (half_t*)alloc((size_t)P * 128 * sizeof(half_t));
    float*  supN   = (float*)alloc((size_t)P * C0 * sizeof(float));
    half_t* offbuf = (half_t*)alloc((size_t)P * OFFPAD * sizeof(half_t));
    half_t* wtbuf  = (half_t*)alloc((size_t)OFFPAD * 9 * 128 * sizeof(half_t));
    float*  scbuf  = (float*)alloc(512 * sizeof(float));
    float*  shbuf  = (float*)alloc(512 * sizeof(float));

    float* out = (float*)d_out;
    const int outN = Bn * C0 * HW;

    zero_f32<<<(outN + 255) / 256, 256, 0, stream>>>(out, outN);
    prep_inputs<<<(P * 32 + 255) / 256, 256, 0, stream>>>(ref, sup, diff32, supN);

    const dim3 cb(256);
    const dim3 grid1(P / 128, 1);

    // ---- block 0: conv1 (26->128) + BN + ReLU
    wt_transform<<<(128 * 9 * 32 + 255) / 256, 256, 0, stream>>>(w1_0, wtbuf, 128, 26, 9, 128, 32);
    bn_fold<<<1, 128, 0, stream>>>(bn1_0, scbuf, shbuf, 128);
    conv_wmma<32><<<grid1, cb, 0, stream>>>(diff32, wtbuf, scbuf, shbuf, nullptr, bufs[0],
                                            128, 128, 9, 1, 1);
    // ---- block 0: downsample 1x1 (26->128) + BN (identity path)
    wt_transform<<<(128 * 1 * 32 + 255) / 256, 256, 0, stream>>>(wd_0, wtbuf, 128, 26, 1, 128, 32);
    bn_fold<<<1, 128, 0, stream>>>(bnd_0, scbuf, shbuf, 128);
    conv_wmma<32><<<grid1, cb, 0, stream>>>(diff32, wtbuf, scbuf, shbuf, nullptr, bufs[1],
                                            128, 128, 1, 1, 0);
    // ---- block 0: conv2 + BN + residual(idn) + ReLU
    wt_transform<<<(128 * 9 * 128 + 255) / 256, 256, 0, stream>>>(w2_0, wtbuf, 128, 128, 9, 128, 128);
    bn_fold<<<1, 128, 0, stream>>>(bn2_0, scbuf, shbuf, 128);
    conv_wmma<128><<<grid1, cb, 0, stream>>>(bufs[0], wtbuf, scbuf, shbuf, bufs[1], bufs[2],
                                             128, 128, 9, 1, 1);

    int cur = 2;
    for (int t = 0; t < 19; ++t) {
        const int i1 = (cur + 1) % 3;
        const int i2 = (cur + 2) % 3;
        wt_transform<<<(128 * 9 * 128 + 255) / 256, 256, 0, stream>>>(
            w1_rest + (size_t)t * 128 * 128 * 9, wtbuf, 128, 128, 9, 128, 128);
        bn_fold<<<1, 128, 0, stream>>>(bn1_rest + (size_t)t * 4 * 128, scbuf, shbuf, 128);
        conv_wmma<128><<<grid1, cb, 0, stream>>>(bufs[cur], wtbuf, scbuf, shbuf, nullptr,
                                                 bufs[i1], 128, 128, 9, 1, 1);
        wt_transform<<<(128 * 9 * 128 + 255) / 256, 256, 0, stream>>>(
            w2_rest + (size_t)t * 128 * 128 * 9, wtbuf, 128, 128, 9, 128, 128);
        bn_fold<<<1, 128, 0, stream>>>(bn2_rest + (size_t)t * 4 * 128, scbuf, shbuf, 128);
        conv_wmma<128><<<grid1, cb, 0, stream>>>(bufs[i1], wtbuf, scbuf, shbuf, bufs[cur],
                                                 bufs[i2], 128, 128, 9, 1, 1);
        cur = i2;
    }

    // ---- 5 dilation branches: offset conv (128->468, dil=pad=d) + deformable conv
    const int dils[5] = {3, 6, 12, 18, 24};
    for (int i = 0; i < 5; ++i) {
        wt_transform<<<(OFFPAD * 9 * 128 + 255) / 256, 256, 0, stream>>>(
            off_ws + (size_t)i * OFFC * 128 * 9, wtbuf, OFFC, 128, 9, OFFPAD, 128);
        conv_wmma<128><<<dim3(P / 128, OFFPAD / 128 + 1), cb, 0, stream>>>(
            bufs[cur], wtbuf, nullptr, nullptr, nullptr, offbuf,
            OFFPAD, OFFPAD, 9, dils[i], 0);
        deform_kernel<<<P / 256, 256, 0, stream>>>(
            supN, offbuf, def_ws + (size_t)i * 26 * 26 * 9, out, dils[i]);
    }
}
